// PointTransformerLayer_17841294147943
// MI455X (gfx1250) — matrix operand order repaired and checked by reading, not compiled, not run on hardware
//
#include <hip/hip_runtime.h>

// ---------------------------------------------------------------------------
// Point Transformer layer for MI455X (gfx1250, wave32, WMMA bf16 16x16x32).
// Kernel 1: phi = PF @ W_phi + b_phi       (pure WMMA GEMM, writes into d_out)
// Kernel 2: per point n (one wave32 each), M=16 neighbor tile:
//   hidden = diff @ W_d1 + b_d1                       (WMMA, K padded to 32)
//   delta  = relu(hidden @ W_d2 + b_d2)               (WMMA)
//   alpha  = NF @ W_alpha + b_alpha + delta           (WMMA)
//   beta   = phi - (NF @ W_psi + b_psi) + delta       (WMMA)
//   gamma  = beta @ W_gamma + b_gamma                 (WMMA)
//   rho    = softmax(gamma, axis=-1); out = sum_k rho*alpha
// Staging tiles between C-layout and A-layout are kept in bf16 so reloads are
// pure ds_load_b128 with no conversion VALU (numerically identical: the data
// would be converted to bf16 for the next WMMA anyway).
// ---------------------------------------------------------------------------

typedef __attribute__((ext_vector_type(16))) __bf16       v16bf;
typedef __attribute__((ext_vector_type(8)))  float        v8f;
typedef __attribute__((ext_vector_type(8)))  unsigned int v8u;

#define FDIM 64
#define KNN  16
#define WSTR 68                 // weight LDS column stride (halfwords)
#define MAT_ELEMS (FDIM * WSTR) // 4352 per 64x64 matrix
#define SSTR 72                 // staging row stride (bf16): 144B = 36 dwords
                                // -> 16 lanes x 4 consecutive dwords cover all
                                //    64 banks exactly once (conflict-free)

__device__ __forceinline__ unsigned int pack2bf(float a, float b) {
  unsigned int ua = __builtin_bit_cast(unsigned int, a);
  unsigned int ub = __builtin_bit_cast(unsigned int, b);
  ua += 0x7FFFu + ((ua >> 16) & 1u);            // round-to-nearest-even
  ub += 0x7FFFu + ((ub >> 16) & 1u);
  return (ua >> 16) | (ub & 0xFFFF0000u);
}

__device__ __forceinline__ unsigned short f2bf(float a) {
  unsigned int ua = __builtin_bit_cast(unsigned int, a);
  ua += 0x7FFFu + ((ua >> 16) & 1u);
  return (unsigned short)(ua >> 16);
}

__device__ __forceinline__ v8f splat8(float x) {
  v8f r;
#pragma unroll
  for (int i = 0; i < 8; ++i) r[i] = x;
  return r;
}

// A-fragment (16x32 bf16, MxK) from a row-major f32 row pointer (global).
// Lane layout (ISA 7.12.2): half h: V0..V3 = K 8h..8h+7, V4..V7 = K 8h+16..23.
__device__ __forceinline__ v16bf afrag(const float* row, int kstep, int h) {
  const float* p = row + kstep * 32 + h * 8;
  float4 x0 = *(const float4*)(p);
  float4 x1 = *(const float4*)(p + 4);
  float4 y0 = *(const float4*)(p + 16);
  float4 y1 = *(const float4*)(p + 20);
  v8u u;
  u[0] = pack2bf(x0.x, x0.y); u[1] = pack2bf(x0.z, x0.w);
  u[2] = pack2bf(x1.x, x1.y); u[3] = pack2bf(x1.z, x1.w);
  u[4] = pack2bf(y0.x, y0.y); u[5] = pack2bf(y0.z, y0.w);
  u[6] = pack2bf(y1.x, y1.y); u[7] = pack2bf(y1.z, y1.w);
  return __builtin_bit_cast(v16bf, u);
}

// A-fragment from bf16 staging in LDS: two 16B loads, no conversion.
__device__ __forceinline__ v16bf afrag_st(const unsigned short* row, int kstep,
                                          int h) {
  const unsigned short* p = row + kstep * 32 + h * 8;   // 16B aligned
  uint4 x = *(const uint4*)(p);        // K 8h .. 8h+7
  uint4 y = *(const uint4*)(p + 16);   // K 8h+16 .. 8h+23
  v8u u;
  u[0] = x.x; u[1] = x.y; u[2] = x.z; u[3] = x.w;
  u[4] = y.x; u[5] = y.y; u[6] = y.z; u[7] = y.w;
  return __builtin_bit_cast(v16bf, u);
}

// B-fragment (32x16 bf16, KxN) from LDS weights stored column-major
// (column c at w[c*WSTR], contiguous in K). Lane: col = lane%16,
// K = 8*(lane/16) + 0..7 and 16 + 8*(lane/16) + 0..7 -> two 16B runs.
__device__ __forceinline__ v16bf bfrag(const unsigned short* w, int tile,
                                       int kstep, int lane) {
  int c = tile * 16 + (lane & 15);
  int h = lane >> 4;
  const unsigned short* p = w + c * WSTR + kstep * 32 + h * 8;   // 8B aligned
  const uint2* q = (const uint2*)p;
  uint2 a = q[0], b = q[1], cc = q[4], d = q[5];
  v8u u;
  u[0] = a.x;  u[1] = a.y;  u[2] = b.x;  u[3] = b.y;
  u[4] = cc.x; u[5] = cc.y; u[6] = d.x;  u[7] = d.y;
  return __builtin_bit_cast(v16bf, u);
}

__device__ __forceinline__ v8f wmma_bf(v16bf a, v16bf b, v8f c) {
  return __builtin_amdgcn_wmma_f32_16x16x32_bf16(false, a, false, b,
                                                 (short)0, c, false, false);
}

// Store a C-tile (VGPR v -> row v+8h, col = 16*tile + lane%16) to bf16 staging.
__device__ __forceinline__ void cstore_bf(unsigned short* st, int tile,
                                          int lane, v8f acc) {
  int c = tile * 16 + (lane & 15);
  int h = lane >> 4;
#pragma unroll
  for (int v = 0; v < 8; ++v) st[(v + 8 * h) * SSTR + c] = f2bf(acc[v]);
}

// ---------------------------------------------------------------------------
// Kernel 1: phi = points_features @ W_phi + b_phi  (written into `out`)
// ---------------------------------------------------------------------------
__global__ void __launch_bounds__(128) phi_gemm_kernel(
    const float* __restrict__ pf, const float* __restrict__ Wphi,
    const float* __restrict__ bphi, float* __restrict__ phiOut, int nPoints) {
  __shared__ unsigned short ldsW[MAT_ELEMS];
  int tid = threadIdx.x;
  for (int i = tid; i < FDIM * FDIM; i += 128) {
    int k = i >> 6, c = i & 63;
    ldsW[c * WSTR + k] = f2bf(Wphi[i]);   // column-major, pre-swizzled
  }
  __syncthreads();

  int lane = tid & 31, wv = tid >> 5;
  int li = lane & 15, h = lane >> 4;
  int nTiles = (nPoints + 15) >> 4;
  int tile = blockIdx.x * 4 + wv;
  if (tile >= nTiles) return;                       // wave-uniform

  int row = tile * 16 + li;
  if (row >= nPoints) row = nPoints - 1;
  const float* rp = pf + (size_t)row * FDIM;
  v16bf a0 = afrag(rp, 0, h);
  v16bf a1 = afrag(rp, 1, h);

  bool fullTile = (tile * 16 + 16 <= nPoints);      // wave-uniform branch

#pragma unroll
  for (int t = 0; t < 4; ++t) {
    int c = t * 16 + li;
    v8f acc = splat8(bphi[c]);
    acc = wmma_bf(a0, bfrag(ldsW, t, 0, lane), acc);
    acc = wmma_bf(a1, bfrag(ldsW, t, 1, lane), acc);
    float* colp = phiOut + (size_t)(tile * 16 + 8 * h) * FDIM + c;
    if (fullTile) {
#pragma unroll
      for (int v = 0; v < 8; ++v) colp[(size_t)v * FDIM] = acc[v];
    } else {
#pragma unroll
      for (int v = 0; v < 8; ++v) {
        int m = tile * 16 + v + 8 * h;
        if (m < nPoints) phiOut[(size_t)m * FDIM + c] = acc[v];
      }
    }
  }
}

// ---------------------------------------------------------------------------
// Kernel 2: fused per-point transformer. One wave32 per point, grid-stride.
// Reads phi from `out`, overwrites the same row with the final result.
// ---------------------------------------------------------------------------
__global__ void __launch_bounds__(128) pt_layer_kernel(
    const float* __restrict__ pxyz,  const float* __restrict__ nxyz,
    const float* __restrict__ nfeat,
    const float* __restrict__ Wpsi,  const float* __restrict__ bpsi,
    const float* __restrict__ Walp,  const float* __restrict__ balp,
    const float* __restrict__ Wgam,  const float* __restrict__ bgam,
    const float* __restrict__ Wd1,   const float* __restrict__ bd1,
    const float* __restrict__ Wd2,   const float* __restrict__ bd2,
    float* out, int nPoints, int nWaves) {
  __shared__ unsigned short ldsW[5 * MAT_ELEMS];    // psi, alpha, gamma, d2, d1
  __shared__ unsigned short stageB[4][16 * SSTR];   // per-wave bf16 staging

  int tid = threadIdx.x;
  for (int i = tid; i < 5 * FDIM * FDIM; i += 128) {
    int m = i >> 12, e = i & 4095;
    int k = e >> 6, c = e & 63;
    float v;
    if      (m == 0) v = Wpsi[e];
    else if (m == 1) v = Walp[e];
    else if (m == 2) v = Wgam[e];
    else if (m == 3) v = Wd2[e];
    else             v = (k < 3) ? Wd1[e] : 0.0f;  // W_d1 zero-padded to K=64
    ldsW[m * MAT_ELEMS + c * WSTR + k] = f2bf(v);
  }
  __syncthreads();

  int lane = tid & 31, wv = tid >> 5;
  int li = lane & 15, h = lane >> 4;
  unsigned short* st = &stageB[wv][0];
  const unsigned short* Wpsi_l = ldsW;
  const unsigned short* Wal_l  = ldsW + 1 * MAT_ELEMS;
  const unsigned short* Wga_l  = ldsW + 2 * MAT_ELEMS;
  const unsigned short* Wd2_l  = ldsW + 3 * MAT_ELEMS;
  const unsigned short* Wd1_l  = ldsW + 4 * MAT_ELEMS;

  float bps[4], bal4[4], bga[4], b1[4], b2[4];
#pragma unroll
  for (int t = 0; t < 4; ++t) {
    int c = t * 16 + li;
    bps[t] = bpsi[c]; bal4[t] = balp[c]; bga[t] = bgam[c];
    b1[t]  = bd1[c];  b2[t]   = bd2[c];
  }

  for (int n = blockIdx.x * 4 + wv; n < nPoints; n += nWaves) {
    // Prefetch next point's neighbour-feature tile (4KB, 128B per lane).
    if (n + nWaves < nPoints) {
      const float* nxt = nfeat + (size_t)(n + nWaves) * KNN * FDIM + lane * 32;
      __builtin_prefetch(nxt, 0, 1);
    }

    // A-fragments of neighbours_features[n] : 16 x 64 (M x K)
    const float* nf = nfeat + (size_t)n * KNN * FDIM + (size_t)li * FDIM;
    v16bf aNF0 = afrag(nf, 0, h);
    v16bf aNF1 = afrag(nf, 1, h);

    // diff = p_i - p_j, as a 16x32 A-fragment (K=0..2 valid, rest zero)
    float px = pxyz[n * 3 + 0], py = pxyz[n * 3 + 1], pz = pxyz[n * 3 + 2];
    const float* nx = nxyz + ((size_t)n * KNN + li) * 3;
    float dx = px - nx[0], dy = py - nx[1], dz = pz - nx[2];
    unsigned int p01 = pack2bf(dx, dy), p2 = pack2bf(dz, 0.0f);
    v8u ud;
#pragma unroll
    for (int i = 0; i < 8; ++i) ud[i] = 0u;
    ud[0] = h ? 0u : p01;
    ud[1] = h ? 0u : p2;
    v16bf aD = __builtin_bit_cast(v16bf, ud);

    // hidden = diff @ W_d1 + b_d1  -> bf16 staging (C-layout -> row-major)
#pragma unroll
    for (int t = 0; t < 4; ++t) {
      v8f acc = splat8(b1[t]);
      acc = wmma_bf(aD, bfrag(Wd1_l, t, 0, lane), acc);
      cstore_bf(st, t, lane, acc);
    }
    asm volatile("s_wait_dscnt 0x0" ::: "memory");
    const unsigned short* srow = st + li * SSTR;
    v16bf aH0 = afrag_st(srow, 0, h);
    v16bf aH1 = afrag_st(srow, 1, h);

    // delta = relu(hidden @ W_d2 + b_d2)
    v8f delta[4];
#pragma unroll
    for (int t = 0; t < 4; ++t) {
      v8f acc = splat8(b2[t]);
      acc = wmma_bf(aH0, bfrag(Wd2_l, t, 0, lane), acc);
      acc = wmma_bf(aH1, bfrag(Wd2_l, t, 1, lane), acc);
#pragma unroll
      for (int v = 0; v < 8; ++v) acc[v] = fmaxf(acc[v], 0.0f);
      delta[t] = acc;
    }

    // alpha_out = NF @ W_alpha + b_alpha + delta
    v8f alf[4];
#pragma unroll
    for (int t = 0; t < 4; ++t) {
      v8f acc = splat8(bal4[t]);
      acc = wmma_bf(aNF0, bfrag(Wal_l, t, 0, lane), acc);
      acc = wmma_bf(aNF1, bfrag(Wal_l, t, 1, lane), acc);
      alf[t] = acc + delta[t];
    }

    // beta = phi - (NF @ W_psi + b_psi) + delta  -> bf16 staging
#pragma unroll
    for (int t = 0; t < 4; ++t) {
      v8f acc = splat8(bps[t]);
      acc = wmma_bf(aNF0, bfrag(Wpsi_l, t, 0, lane), acc);
      acc = wmma_bf(aNF1, bfrag(Wpsi_l, t, 1, lane), acc);
      float phic = out[(size_t)n * FDIM + t * 16 + li];  // phi from kernel 1
      v8f beta;
#pragma unroll
      for (int v = 0; v < 8; ++v) beta[v] = phic + delta[t][v] - acc[v];
      cstore_bf(st, t, lane, beta);
    }
    asm volatile("s_wait_dscnt 0x0" ::: "memory");
    v16bf aB0 = afrag_st(srow, 0, h);
    v16bf aB1 = afrag_st(srow, 1, h);

    // gamma = beta @ W_gamma + b_gamma
    v8f g[4];
#pragma unroll
    for (int t = 0; t < 4; ++t) {
      v8f acc = splat8(bga[t]);
      acc = wmma_bf(aB0, bfrag(Wga_l, t, 0, lane), acc);
      acc = wmma_bf(aB1, bfrag(Wga_l, t, 1, lane), acc);
      g[t] = acc;
    }

    // rho = softmax(gamma, axis=-1): row m = v+8h spans 16 lanes x 4 tiles.
    float rm[8], ri[8];
#pragma unroll
    for (int v = 0; v < 8; ++v) {
      float m = fmaxf(fmaxf(g[0][v], g[1][v]), fmaxf(g[2][v], g[3][v]));
      m = fmaxf(m, __shfl_xor(m, 1, 32));
      m = fmaxf(m, __shfl_xor(m, 2, 32));
      m = fmaxf(m, __shfl_xor(m, 4, 32));
      m = fmaxf(m, __shfl_xor(m, 8, 32));
      rm[v] = m;
    }
    v8f e[4];
#pragma unroll
    for (int t = 0; t < 4; ++t)
#pragma unroll
      for (int v = 0; v < 8; ++v) e[t][v] = __expf(g[t][v] - rm[v]);
#pragma unroll
    for (int v = 0; v < 8; ++v) {
      float s = e[0][v] + e[1][v] + e[2][v] + e[3][v];
      s += __shfl_xor(s, 1, 32);
      s += __shfl_xor(s, 2, 32);
      s += __shfl_xor(s, 4, 32);
      s += __shfl_xor(s, 8, 32);
      ri[v] = 1.0f / s;
    }

    // out[n][c] = sum_k rho * alpha  (sum over rows m)
    float o[4];
#pragma unroll
    for (int t = 0; t < 4; ++t) {
      float a = 0.0f;
#pragma unroll
      for (int v = 0; v < 8; ++v) a += e[t][v] * ri[v] * alf[t][v];
      o[t] = a;
    }
#pragma unroll
    for (int t = 0; t < 4; ++t) o[t] += __shfl_xor(o[t], 16, 32);
    if (lane < 16) {
#pragma unroll
      for (int t = 0; t < 4; ++t)
        out[(size_t)n * FDIM + t * 16 + lane] = o[t];
    }
  }
}

// ---------------------------------------------------------------------------
extern "C" void kernel_launch(void* const* d_in, const int* in_sizes, int n_in,
                              void* d_out, int out_size, void* d_ws,
                              size_t ws_size, hipStream_t stream) {
  const float* pxyz  = (const float*)d_in[0];
  const float* pfeat = (const float*)d_in[1];
  const float* nxyz  = (const float*)d_in[2];
  const float* nfeat = (const float*)d_in[3];
  const float* Wphi  = (const float*)d_in[4];
  const float* bphi  = (const float*)d_in[5];
  const float* Wpsi  = (const float*)d_in[6];
  const float* bpsi  = (const float*)d_in[7];
  const float* Walp  = (const float*)d_in[8];
  const float* balp  = (const float*)d_in[9];
  const float* Wgam  = (const float*)d_in[10];
  const float* bgam  = (const float*)d_in[11];
  const float* Wd1   = (const float*)d_in[12];
  const float* bd1   = (const float*)d_in[13];
  const float* Wd2   = (const float*)d_in[14];
  const float* bd2   = (const float*)d_in[15];
  float* out = (float*)d_out;

  int nPoints = in_sizes[0] / 3;              // B*N*3 -> N (B==1)
  int nTiles  = (nPoints + 15) / 16;
  int blocks1 = (nTiles + 3) / 4;             // 4 waves / block, 1 tile / wave
  phi_gemm_kernel<<<blocks1, 128, 0, stream>>>(pfeat, Wphi, bphi, out, nPoints);

  int blocks2 = 512;                          // 2048 waves, grid-stride points
  pt_layer_kernel<<<blocks2, 128, 0, stream>>>(
      pxyz, nxyz, nfeat, Wpsi, bpsi, Walp, balp, Wgam, bgam, Wd1, bd1, Wd2, bd2,
      out, nPoints, blocks2 * 4);
}
